// MYAttention_1262720385465
// MI455X (gfx1250) — matrix operand order repaired and checked by reading, not compiled, not run on hardware
//
#include <hip/hip_runtime.h>
#include <stdint.h>

// ---- problem constants (match reference) ----
#define B_DIM 64
#define N_DIM 2048
#define C_DIM 128
#define F_DIM 1024   // also H
#define A_DIM 512

typedef __attribute__((ext_vector_type(16))) __bf16 v16bf;
typedef __attribute__((ext_vector_type(8)))  float  v8f;

// RNE float->bf16 pack of two values into one dword (lo = first)
__device__ __forceinline__ unsigned packbf2(float a, float b) {
  unsigned ua = __float_as_uint(a), ub = __float_as_uint(b);
  ua = (ua + 0x7FFFu + ((ua >> 16) & 1u)) >> 16;
  ub = (ub + 0x7FFFu + ((ub >> 16) & 1u)) >> 16;
  return ua | (ub << 16);
}

// ---- one-time weight conversion: f32 -> bf16 ----
__global__ void cvt_bf16_kernel(const float* __restrict__ src,
                                unsigned short* __restrict__ dst, int n) {
  int i = (blockIdx.x * blockDim.x + threadIdx.x) * 2;
  if (i < n) {
    *(unsigned*)(dst + i) = packbf2(src[i], src[i + 1]);
  }
}

// ---- h[b,a] = key[b,:]·Wh[a,:] + bh[a] + bv[a]  (bv folded in) ----
__global__ void h_bias_kernel(const float* __restrict__ key,
                              const float* __restrict__ Wh,
                              const float* __restrict__ bh,
                              const float* __restrict__ bv,
                              float* __restrict__ hb) {
  int b = blockIdx.x;
  int a = blockIdx.y * blockDim.x + threadIdx.x;       // A = 512, grid.y = 2
  const float4* kp = (const float4*)(key + (size_t)b * F_DIM);
  const float4* wp = (const float4*)(Wh + (size_t)a * F_DIM);
  float acc = bh[a] + bv[a];
  for (int i = 0; i < F_DIM / 4; ++i) {
    float4 k4 = kp[i], w4 = wp[i];
    acc += k4.x * w4.x + k4.y * w4.y + k4.z * w4.z + k4.w * w4.w;
  }
  hb[(size_t)b * A_DIM + a] = acc;
}

// ---- WMMA score kernel: scores[b,l] = wa · tanh(h_b + x_l @ W^T) ----
#define MTILE 32
#define KPH 512
#define LDS_STRIDE 520   // 512 + 8 pad (bf16 elems) -> conflict-free frag loads

__global__ void __launch_bounds__(256) scores_kernel(
    const float* __restrict__ x, const unsigned short* __restrict__ Wb,
    const float* __restrict__ hb, const float* __restrict__ wa,
    float* __restrict__ scores, int Lrows) {
  __shared__ __align__(16) unsigned short tileb[MTILE * LDS_STRIDE];
  __shared__ float sred[MTILE];

  const int tid  = threadIdx.x;
  const int lane = tid & 31, wave = tid >> 5;
  const int wm = wave & 1, wn = wave >> 1;   // 2 M-subtiles x 4 N-groups
  const int hi = lane >> 4, lm = lane & 15;
  const int b = blockIdx.y;
  const int row0 = blockIdx.x * MTILE;
  const float* xblk = x + ((size_t)b * Lrows + row0) * F_DIM;

  if (tid < MTILE) sred[tid] = 0.0f;

  v8f zero;
  #pragma unroll
  for (int j = 0; j < 8; ++j) zero[j] = 0.0f;
  v8f acc[8];
  #pragma unroll
  for (int t = 0; t < 8; ++t) acc[t] = zero;

  const int mrow = wm * 16 + lm;

  for (int ph = 0; ph < F_DIM / KPH; ++ph) {
    __syncthreads();
    // stage MTILE x KPH f32 -> bf16 LDS (4096 float4 / 256 threads)
    const float* xsrc = xblk + ph * KPH;
    for (int i = tid; i < MTILE * (KPH / 4); i += 256) {
      int r = i >> 7;          // / (KPH/4)
      int c4 = i & 127;
      float4 v = *(const float4*)(xsrc + (size_t)r * F_DIM + c4 * 4);
      *(uint2*)(&tileb[r * LDS_STRIDE + c4 * 4]) =
          make_uint2(packbf2(v.x, v.y), packbf2(v.z, v.w));
    }
    __syncthreads();

    for (int kb = 0; kb < KPH; kb += 32) {
      // A fragment (16x32 bf16): lower lanes K {0-7,16-23}, upper +8
      union { v16bf v; uint4 q[2]; } afrag;
      const unsigned short* ap = &tileb[mrow * LDS_STRIDE + kb + hi * 8];
      afrag.q[0] = *(const uint4*)(ap);
      afrag.q[1] = *(const uint4*)(ap + 16);

      const int kglob = ph * KPH + kb;
      #pragma unroll
      for (int t = 0; t < 8; ++t) {
        int a_col = wn * 128 + t * 16 + lm;
        // B fragment (32x16 bf16): lane = column, lower lanes K {0-7,16-23}, upper +8
        union { v16bf v; uint4 q[2]; } bfrag;
        const uint4* bp =
            (const uint4*)(Wb + (size_t)a_col * F_DIM + kglob + hi * 8);
        bfrag.q[0] = bp[0];
        bfrag.q[1] = bp[2];
        acc[t] = __builtin_amdgcn_wmma_f32_16x16x32_bf16(
            false, afrag.v, false, bfrag.v, (short)0, acc[t], false, false);
      }
    }
  }

  // epilogue: partial_m += tanh(v + h[a]) * wa[a], reduce rows via LDS atomics
  float partial[8];
  #pragma unroll
  for (int r = 0; r < 8; ++r) partial[r] = 0.0f;
  const float* hrow = hb + (size_t)b * A_DIM;
  #pragma unroll
  for (int t = 0; t < 8; ++t) {
    int a_col = wn * 128 + t * 16 + lm;
    float hv = hrow[a_col];
    float wv = wa[a_col];
    #pragma unroll
    for (int r = 0; r < 8; ++r)
      partial[r] += tanhf(acc[t][r] + hv) * wv;   // C layout: M = r + hi*8
  }
  const int rbase = wm * 16 + hi * 8;
  #pragma unroll
  for (int r = 0; r < 8; ++r) atomicAdd(&sred[rbase + r], partial[r]);
  __syncthreads();
  if (tid < MTILE) scores[(size_t)b * Lrows + row0 + tid] = sred[tid];
}

// ---- in-place softmax over L per batch ----
__global__ void softmax_kernel(float* __restrict__ s, int L) {
  __shared__ float red[256];
  int b = blockIdx.x, tid = threadIdx.x;
  float* p = s + (size_t)b * L;
  float mx = -3.4e38f;
  for (int i = tid; i < L; i += 256) mx = fmaxf(mx, p[i]);
  red[tid] = mx; __syncthreads();
  for (int w = 128; w > 0; w >>= 1) {
    if (tid < w) red[tid] = fmaxf(red[tid], red[tid + w]);
    __syncthreads();
  }
  mx = red[0]; __syncthreads();
  float sum = 0.0f;
  for (int i = tid; i < L; i += 256) { float e = __expf(p[i] - mx); p[i] = e; sum += e; }
  red[tid] = sum; __syncthreads();
  for (int w = 128; w > 0; w >>= 1) {
    if (tid < w) red[tid] += red[tid + w];
    __syncthreads();
  }
  float inv = 1.0f / red[0];
  for (int i = tid; i < L; i += 256) p[i] *= inv;
}

// ---- att[b,f] += sum_l alpha[b,l] * x[b,l,f] (L chunked, atomic combine) ----
__global__ void wsum_kernel(const float* __restrict__ x,
                            const float* __restrict__ alpha,
                            float* __restrict__ att, int L, int chunk) {
  int b = blockIdx.x;
  int f = blockIdx.y * 256 + threadIdx.x;
  int l0 = blockIdx.z * chunk;
  const float* xp = x + ((size_t)b * L + l0) * F_DIM + f;
  const float* ap = alpha + (size_t)b * L + l0;
  float acc = 0.0f;
  for (int l = 0; l < chunk; ++l) acc += ap[l] * xp[(size_t)l * F_DIM];
  atomicAdd(&att[(size_t)b * F_DIM + f], acc);
}

// ---- gate = sigmoid([att1,key,att2] @ Wg^T + bg); out = blend ----
__global__ void gate_kernel(const float* __restrict__ att1,
                            const float* __restrict__ key,
                            const float* __restrict__ att2,
                            const float* __restrict__ Wg,
                            const float* __restrict__ bg,
                            float* __restrict__ out) {
  __shared__ __align__(16) float gin[3 * F_DIM];
  int b = blockIdx.x, tid = threadIdx.x;
  for (int i = tid; i < F_DIM; i += 256) {
    gin[i]             = att1[(size_t)b * F_DIM + i];
    gin[F_DIM + i]     = key [(size_t)b * F_DIM + i];
    gin[2 * F_DIM + i] = att2[(size_t)b * F_DIM + i];
  }
  __syncthreads();
  for (int ff = 0; ff < 4; ++ff) {
    int f = ff * 256 + tid;
    const float4* wrow = (const float4*)(Wg + (size_t)f * (3 * F_DIM));
    const float4* g4 = (const float4*)gin;
    float acc = bg[f];
    for (int k = 0; k < (3 * F_DIM) / 4; ++k) {
      float4 w = wrow[k], g = g4[k];
      acc += w.x * g.x + w.y * g.y + w.z * g.z + w.w * g.w;
    }
    float gate = 1.0f / (1.0f + __expf(-acc));
    out[(size_t)b * F_DIM + f] = gate * gin[f] + (1.0f - gate) * gin[2 * F_DIM + f];
  }
}

extern "C" void kernel_launch(void* const* d_in, const int* in_sizes, int n_in,
                              void* d_out, int out_size, void* d_ws, size_t ws_size,
                              hipStream_t stream) {
  const float* feats  = (const float*)d_in[0];
  const float* cfeats = (const float*)d_in[1];
  const float* key    = (const float*)d_in[2];
  const float* Wv     = (const float*)d_in[3];
  const float* bv     = (const float*)d_in[4];
  const float* Wh     = (const float*)d_in[5];
  const float* bh     = (const float*)d_in[6];
  const float* wa     = (const float*)d_in[7];
  const float* Wv2    = (const float*)d_in[8];
  const float* bv2    = (const float*)d_in[9];
  const float* Wh2    = (const float*)d_in[10];
  const float* bh2    = (const float*)d_in[11];
  const float* wa2    = (const float*)d_in[12];
  const float* Wg     = (const float*)d_in[13];
  const float* bg     = (const float*)d_in[14];
  float* out = (float*)d_out;

  char* ws = (char*)d_ws;
  unsigned short* Wvb  = (unsigned short*)(ws);                 // 1 MB
  unsigned short* Wv2b = (unsigned short*)(ws + (1u << 20));    // 1 MB
  float* hb1  = (float*)(ws + (2u << 20));                      // 128 KB
  float* hb2  = (float*)(ws + 2228224u);                        // 128 KB
  float* sc1  = (float*)(ws + 2359296u);                        // 512 KB
  float* sc2  = (float*)(ws + 2883584u);                        // 32 KB
  float* att1 = (float*)(ws + 2916352u);                        // 256 KB
  float* att2 = (float*)(ws + 3178496u);                        // 256 KB

  const int nW = A_DIM * F_DIM;   // 524288
  cvt_bf16_kernel<<<nW / 512, 256, 0, stream>>>(Wv,  Wvb,  nW);
  cvt_bf16_kernel<<<nW / 512, 256, 0, stream>>>(Wv2, Wv2b, nW);

  h_bias_kernel<<<dim3(B_DIM, A_DIM / 256), 256, 0, stream>>>(key, Wh,  bh,  bv,  hb1);
  h_bias_kernel<<<dim3(B_DIM, A_DIM / 256), 256, 0, stream>>>(key, Wh2, bh2, bv2, hb2);

  scores_kernel<<<dim3(N_DIM / MTILE, B_DIM), 256, 0, stream>>>(feats,  Wvb,  hb1, wa,  sc1, N_DIM);
  scores_kernel<<<dim3(C_DIM / MTILE, B_DIM), 256, 0, stream>>>(cfeats, Wv2b, hb2, wa2, sc2, C_DIM);

  softmax_kernel<<<B_DIM, 256, 0, stream>>>(sc1, N_DIM);
  softmax_kernel<<<B_DIM, 256, 0, stream>>>(sc2, C_DIM);

  hipMemsetAsync(att1, 0, B_DIM * F_DIM * sizeof(float), stream);
  hipMemsetAsync(att2, 0, B_DIM * F_DIM * sizeof(float), stream);
  wsum_kernel<<<dim3(B_DIM, F_DIM / 256, 4), 256, 0, stream>>>(feats,  sc1, att1, N_DIM, 512);
  wsum_kernel<<<dim3(B_DIM, F_DIM / 256, 1), 256, 0, stream>>>(cfeats, sc2, att2, C_DIM, 128);

  gate_kernel<<<B_DIM, 256, 0, stream>>>(att1, key, att2, Wg, bg, out);
}
